// GCNWithAttention_sign_44152263803378
// MI455X (gfx1250) — matrix operand order repaired and checked by reading, not compiled, not run on hardware
//
#include <hip/hip_runtime.h>
#include <hip/hip_bf16.h>

// ---------- CDNA5 WMMA vector types ----------
typedef _Float16 half16 __attribute__((ext_vector_type(16)));
typedef _Float16 half8  __attribute__((ext_vector_type(8)));
typedef float    f32x8  __attribute__((ext_vector_type(8)));
typedef float    f32x4  __attribute__((ext_vector_type(4)));

union AFrag { half16 v; half8 h[2]; };

#define N_NODES 8192
#define KNBR    32
#define DFEAT   128
#define HID     256
#define ROWS_PER_BLOCK 64   // 2 nodes * 32 neighbors

// 1 = use inline-asm gfx1250 async global->LDS DMA for the gather stage.
#define USE_ASYNC_ASM 1

typedef __attribute__((address_space(3))) char lds_char;

static __device__ __forceinline__ void wait_asynccnt0() {
#if __has_builtin(__builtin_amdgcn_s_wait_asynccnt)
    __builtin_amdgcn_s_wait_asynccnt(0);
#else
    asm volatile("s_wait_asynccnt 0x0" ::: "memory");
#endif
}

// Issue one 16B async DMA from global memory into LDS (ASYNCcnt-tracked).
static __device__ __forceinline__ void async_g2l_b128(const void* gsrc, void* ldst) {
    unsigned long long ga = (unsigned long long)gsrc;           // flat == global VA
    unsigned loff = (unsigned)(unsigned long long)(lds_char*)ldst; // LDS byte offset
    asm volatile("global_load_async_to_lds_b128 %0, %1, off"
                 :: "v"(loff), "v"(ga)
                 : "memory");
}

// ---------------------------------------------------------------------------
// Convert x (f32) -> f16
// ---------------------------------------------------------------------------
__global__ void cvt_x_kernel(const float* __restrict__ x, _Float16* __restrict__ xh, int n) {
    int i = blockIdx.x * 256 + threadIdx.x;
    if (i < n) xh[i] = (_Float16)x[i];
}

// ---------------------------------------------------------------------------
// Repack W1/W2 (f32 [256][256], row-major K x N) into CDNA5 B-fragment layout:
//   Wp[((nt*8 + k0)*32 + lane)*16 + e] = (f16) W[k*256 + n]
//   with n = nt*16 + (lane&15),  k = k0*32 + (lane>>4)*16 + e
// ---------------------------------------------------------------------------
__global__ void prep_weights_kernel(const float* __restrict__ W1,
                                    const float* __restrict__ W2,
                                    _Float16* __restrict__ Wp1,
                                    _Float16* __restrict__ Wp2) {
    int idx = blockIdx.x * 256 + threadIdx.x;      // 0 .. 131071
    int mtx = idx >> 16;
    int r   = idx & 65535;
    int e    = r & 15;
    int lane = (r >> 4) & 31;
    int k0   = (r >> 9) & 7;
    int nt   = r >> 12;                            // 0..15
    int k = k0 * 32 + (lane >> 4) * 16 + e;
    int n = nt * 16 + (lane & 15);
    const float* W  = mtx ? W2  : W1;
    _Float16*    Wp = mtx ? Wp2 : Wp1;
    Wp[r] = (_Float16)W[k * HID + n];
}

// ---------------------------------------------------------------------------
// Zero the 8192x8192 pairwise output (268 MB) with non-temporal wide stores.
// ---------------------------------------------------------------------------
__global__ void zero_pair_kernel(f32x4* __restrict__ p, unsigned long long n4) {
    unsigned long long i = ((unsigned long long)blockIdx.x * blockDim.x + threadIdx.x) * 2ull;
    f32x4 z = {0.f, 0.f, 0.f, 0.f};
    if (i + 1 < n4) {
        __builtin_nontemporal_store(z, &p[i]);
        __builtin_nontemporal_store(z, &p[i + 1]);
    }
}

// ---------------------------------------------------------------------------
// One MLP layer on 64 rows: sDst = act(sSrc(64x256,f16) @ W(256x256) + bias)
// ---------------------------------------------------------------------------
template <bool RELU>
static __device__ __forceinline__ void mlp_layer(const _Float16* __restrict__ sSrc,
                                                 _Float16* __restrict__ sDst,
                                                 const _Float16* __restrict__ Wp,
                                                 const float* __restrict__ bias) {
    const int tid  = threadIdx.x;
    const int wv   = tid >> 5;
    const int lane = tid & 31;
    const int m    = lane & 15;   // A row within tile / D column
    const int hi   = lane >> 4;

    f32x8 acc[4][2];
#pragma unroll
    for (int mt = 0; mt < 4; ++mt)
#pragma unroll
        for (int nn = 0; nn < 2; ++nn)
#pragma unroll
            for (int r = 0; r < 8; ++r) acc[mt][nn][r] = 0.f;

#pragma unroll
    for (int k0 = 0; k0 < 8; ++k0) {
        // B fragments for the wave's two N-tiles: 32 contiguous bytes per lane
        half16 bf0 = *(const half16*)(Wp + ((((2 * wv + 0) * 8 + k0) * 32 + lane) << 4));
        half16 bf1 = *(const half16*)(Wp + ((((2 * wv + 1) * 8 + k0) * 32 + lane) << 4));
#pragma unroll
        for (int mt = 0; mt < 4; ++mt) {
            // A fragment, contiguous k = 16*hi + e (matches Wp packing): two 16B LDS loads
            const _Float16* arow = sSrc + (mt * 16 + m) * HID + k0 * 32 + hi * 16;
            AFrag af;
            af.h[0] = *(const half8*)(arow);
            af.h[1] = *(const half8*)(arow + 8);
            acc[mt][0] = __builtin_amdgcn_wmma_f32_16x16x32_f16(
                false, af.v, false, bf0, (short)0, acc[mt][0], false, false);
            acc[mt][1] = __builtin_amdgcn_wmma_f32_16x16x32_f16(
                false, af.v, false, bf1, (short)0, acc[mt][1], false, false);
        }
    }

    // Epilogue: bias + activation, write f16 row-major into sDst
#pragma unroll
    for (int nn = 0; nn < 2; ++nn) {
        const int col = (2 * wv + nn) * 16 + m;
        const float bv = bias[col];
#pragma unroll
        for (int mt = 0; mt < 4; ++mt)
#pragma unroll
            for (int r = 0; r < 8; ++r) {
                float v = acc[mt][nn][r] + bv;
                if (RELU) v = fmaxf(v, 0.f);
                sDst[(mt * 16 + hi * 8 + r) * HID + col] = (_Float16)v;
            }
    }
}

// ---------------------------------------------------------------------------
// Main kernel: 2 nodes (64 MLP rows) per block of 256 threads (8 wave32).
// ---------------------------------------------------------------------------
__global__ void __launch_bounds__(256, 2)
gcn_main_kernel(const _Float16* __restrict__ xh,
                const int* __restrict__ nbrs,
                const float* __restrict__ t,
                const float* __restrict__ ehat,
                const _Float16* __restrict__ Wp1, const float* __restrict__ b1,
                const _Float16* __restrict__ Wp2, const float* __restrict__ b2,
                const float* __restrict__ W3, const float* __restrict__ b3,
                const float* __restrict__ pb,
                float* __restrict__ outY, float* __restrict__ pair) {
    __shared__ _Float16 sA[ROWS_PER_BLOCK * HID];   // 32 KB
    __shared__ _Float16 sH[ROWS_PER_BLOCK * HID];   // 32 KB

    const int tid = threadIdx.x;
    const int nodeBase = blockIdx.x * 2;

    // ---- Stage 1: gather [x_i | x_j] rows into sA (f16) ----
    {
        const int r   = tid >> 2;      // row 0..63
        const int qtr = tid & 3;       // 64-half chunk
        const int node = nodeBase + (r >> 5);
        const int j    = r & 31;
        const int cj   = nbrs[node * 33 + 1 + j];
        const _Float16* src = (qtr < 2) ? (xh + node * DFEAT + qtr * 64)
                                        : (xh + cj * DFEAT + (qtr - 2) * 64);
        _Float16* dst = sA + r * HID + qtr * 64;
#if USE_ASYNC_ASM
        // gfx1250 async DMA into LDS (no VGPR staging, ASYNCcnt-tracked)
#pragma unroll
        for (int u = 0; u < 8; ++u)
            async_g2l_b128(src + u * 8, dst + u * 8);
        wait_asynccnt0();
#else
        const uint4* s4 = (const uint4*)src;
        uint4* d4 = (uint4*)dst;
#pragma unroll
        for (int u = 0; u < 8; ++u) d4[u] = s4[u];
#endif
    }
    __syncthreads();

    // ---- Stage 2/3: two WMMA layers (ping-pong LDS) ----
    mlp_layer<true>(sA, sH, Wp1, b1);
    __syncthreads();
    mlp_layer<true>(sH, sA, Wp2, b2);
    __syncthreads();

    // ---- Stage 4: L3 dot products; partials into sH (reused as f32) ----
    float* sP = (float*)sH;
    {
        const int row  = tid & 63;
        const int part = tid >> 6;     // 4 chunks of 64 cols
        const _Float16* h2 = sA + row * HID + part * 64;
        const float* w3 = W3 + part * 64;
        float acc = 0.f;
#pragma unroll
        for (int c = 0; c < 64; ++c) acc += (float)h2[c] * w3[c];
        sP[part * 64 + row] = acc;
    }
    __syncthreads();

    // ---- Stage 5: softmax over K=32 per node, outputs + scatter ----
    if (tid < 64) {                    // waves 0 and 1, fully active
        const int q    = tid >> 5;     // node within block
        const int lane = tid & 31;     // neighbor slot
        const int node = nodeBase + q;
        const int r    = tid;
        const float mval = sP[r] + sP[64 + r] + sP[128 + r] + sP[192 + r] + b3[0];
        const float bscale = pb[0];
        float s = bscale * fabsf(mval);
        float mx = s;
#pragma unroll
        for (int off = 16; off > 0; off >>= 1) mx = fmaxf(mx, __shfl_xor(mx, off, 32));
        float e = __expf(s - mx);
        float se = e;
#pragma unroll
        for (int off = 16; off > 0; off >>= 1) se += __shfl_xor(se, off, 32);
        const float w = mval * (e / se);

        const int cj = nbrs[node * 33 + 1 + lane];
        float y = (t[cj] - ehat[cj]) * w;
#pragma unroll
        for (int off = 16; off > 0; off >>= 1) y += __shfl_xor(y, off, 32);

        pair[(size_t)node * N_NODES + cj] = w;          // scatter
        if (lane == 0) {
            outY[node] = y;
            pair[(size_t)node * N_NODES + node] = 0.f;  // diagonal (after scatter: same-wave store order)
        }
    }
}

// ---------------------------------------------------------------------------
extern "C" void kernel_launch(void* const* d_in, const int* in_sizes, int n_in,
                              void* d_out, int out_size, void* d_ws, size_t ws_size,
                              hipStream_t stream) {
    const float* x    = (const float*)d_in[0];
    const int*   nbrs = (const int*)d_in[1];
    const float* t    = (const float*)d_in[2];
    const float* ehat = (const float*)d_in[3];
    const float* W1   = (const float*)d_in[4];
    const float* b1   = (const float*)d_in[5];
    const float* W2   = (const float*)d_in[6];
    const float* b2   = (const float*)d_in[7];
    const float* W3   = (const float*)d_in[8];
    const float* b3   = (const float*)d_in[9];
    const float* pb   = (const float*)d_in[10];

    float* outY = (float*)d_out;
    float* pair = (float*)d_out + N_NODES;

    char* ws = (char*)d_ws;
    _Float16* xh  = (_Float16*)ws;                        // 2 MB
    _Float16* Wp1 = (_Float16*)(ws + (2u << 20));         // 128 KB
    _Float16* Wp2 = Wp1 + HID * HID;                      // 128 KB

    cvt_x_kernel<<<(N_NODES * DFEAT) / 256, 256, 0, stream>>>(x, xh, N_NODES * DFEAT);
    prep_weights_kernel<<<(2 * HID * HID) / 256, 256, 0, stream>>>(W1, W2, Wp1, Wp2);

    unsigned long long n4 = (unsigned long long)N_NODES * N_NODES / 4;   // float4 count
    zero_pair_kernel<<<(unsigned)(n4 / 512), 256, 0, stream>>>((f32x4*)pair, n4);

    gcn_main_kernel<<<N_NODES / 2, 256, 0, stream>>>(xh, nbrs, t, ehat,
                                                     Wp1, b1, Wp2, b2, W3, b3, pb,
                                                     outY, pair);
}
